// GAT_30219389894764
// MI455X (gfx1250) — matrix operand order repaired and checked by reading, not compiled, not run on hardware
//
#include <hip/hip_runtime.h>
#include <hip/hip_bf16.h>
#include <stdint.h>

#define N_NODES  100000
#define E_EDGES  3200000
#define F_IN     512
#define H1C      32
#define C_OUT    16
#define NEG_SLOPE 0.2f

typedef __bf16 bf16_t;
typedef __bf16 v16bf __attribute__((ext_vector_type(16)));
typedef float  v8f   __attribute__((ext_vector_type(8)));

// ---------------- CDNA5 primitives ----------------

__device__ __forceinline__ v8f wmma_bf16(v16bf a, v16bf b, v8f c) {
  // D = A(16x32 bf16) * B(32x16 bf16) + C(16x16 f32)  -> v_wmma_f32_16x16x32_bf16
  return __builtin_amdgcn_wmma_f32_16x16x32_bf16(false, a, false, b, (short)0, c,
                                                 false, false);
}

__device__ __forceinline__ void atomic_add_f32(float* p, float v) {
  // native FP32 atomic add, device scope (no return -> STOREcnt, drained at s_endpgm)
  asm volatile("global_atomic_add_f32 %0, %1, off scope:SCOPE_DEV"
               :: "v"((unsigned long long)(uintptr_t)p), "v"(v)
               : "memory");
}

// order-preserving float <-> uint mapping so segment-max can use global_atomic_max_u32
__device__ __forceinline__ unsigned f2ord(float f) {
  unsigned u = __float_as_uint(f);
  return (u & 0x80000000u) ? ~u : (u | 0x80000000u);
}
__device__ __forceinline__ float ord2f(unsigned u) {
  return (u & 0x80000000u) ? __uint_as_float(u & 0x7fffffffu) : __uint_as_float(~u);
}
#define ORD_NEG_INF 0x007FFFFFu   // f2ord(-inf)

// ---------------- utility kernels ----------------

__global__ void fill_u32(unsigned* __restrict__ p, unsigned v, long long n) {
  long long i = blockIdx.x * (long long)blockDim.x + threadIdx.x;
  long long s = (long long)blockDim.x * gridDim.x;
  for (; i < n; i += s) p[i] = v;
}

// Pack W (row-major K x Ncols, fp32) into WMMA B-operand lane layout, bf16 hi/lo split.
// B 32x16 bf16 packing used here (contiguous k, not AMD's table order): lanes 0-15: elem e -> (K=e, N=lane); lanes 16-31: elem e -> (K=16+e, N=lane-16)
// Storage: [chunk][tile][lane][e], 16 bf16 per (chunk,tile,lane) record.
__global__ void pack_weights(const float* __restrict__ W1, const float* __restrict__ W2,
                             bf16_t* __restrict__ w1hi, bf16_t* __restrict__ w1lo,
                             bf16_t* __restrict__ w2hi, bf16_t* __restrict__ w2lo) {
  int tid = threadIdx.x;                 // 1024 threads = 16 chunks * 2 tiles * 32 lanes
  {
    int lane = tid & 31, t = (tid >> 5) & 1, c = tid >> 6;
    int half = lane >> 4, n = t * 16 + (lane & 15);
    #pragma unroll
    for (int e = 0; e < 16; ++e) {
      int K = c * 32 + 16 * half + e;
      float v = W1[K * H1C + n];
      bf16_t hi = (bf16_t)v;
      w1hi[tid * 16 + e] = hi;
      w1lo[tid * 16 + e] = (bf16_t)(v - (float)hi);
    }
  }
  if (tid < 32) {                         // layer 2: 1 chunk * 1 tile * 32 lanes
    int lane = tid, half = lane >> 4, n = lane & 15;
    #pragma unroll
    for (int e = 0; e < 16; ++e) {
      int K = 16 * half + e;
      float v = W2[K * C_OUT + n];
      bf16_t hi = (bf16_t)v;
      w2hi[tid * 16 + e] = hi;
      w2lo[tid * 16 + e] = (bf16_t)(v - (float)hi);
    }
  }
}

// ---------------- WMMA GEMM: D[N_NODES x NCOLS] = A[N_NODES x KDIM] * B ----------------
// One wave per 16-row tile. 3-term bf16 split per tile: ahi*bhi + ahi*blo + alo*bhi.
template <int KDIM, int NCOLS>
__global__ __launch_bounds__(64) void gemm_wmma(const float* __restrict__ A,
                                                const bf16_t* __restrict__ Bhi,
                                                const bf16_t* __restrict__ Blo,
                                                float* __restrict__ D) {
  constexpr int CHUNKS = KDIM / 32;
  constexpr int TILES  = NCOLS / 16;
  int wave = blockIdx.x * 2 + (threadIdx.x >> 5);
  int lane = threadIdx.x & 31;
  int half = lane >> 4, l15 = lane & 15;
  int row0 = wave * 16;
  const float* arow = A + (size_t)(row0 + l15) * KDIM;

  v8f acc[TILES] = {};
  for (int c = 0; c < CHUNKS; ++c) {
    // A-operand packing used here (contiguous k, same as B): lane l (l15 = row), pair p in VGPR p:
    //   K = c*32 + 2p + 16*half, elements (2p, 2p+1) = (K, K+1)
    v16bf ahi, alo;
    #pragma unroll
    for (int p = 0; p < 8; ++p) {
      int k0 = c * 32 + 2 * p + 16 * half;
      float f0 = arow[k0], f1 = arow[k0 + 1];
      bf16_t g0 = (bf16_t)f0, g1 = (bf16_t)f1;
      ahi[2 * p]     = g0;
      ahi[2 * p + 1] = g1;
      alo[2 * p]     = (bf16_t)(f0 - (float)g0);
      alo[2 * p + 1] = (bf16_t)(f1 - (float)g1);
    }
    #pragma unroll
    for (int t = 0; t < TILES; ++t) {
      const v16bf bhi = *reinterpret_cast<const v16bf*>(
          Bhi + ((size_t)(c * TILES + t) * 32 + lane) * 16);
      const v16bf blo = *reinterpret_cast<const v16bf*>(
          Blo + ((size_t)(c * TILES + t) * 32 + lane) * 16);
      acc[t] = wmma_bf16(ahi, bhi, acc[t]);
      acc[t] = wmma_bf16(ahi, blo, acc[t]);
      acc[t] = wmma_bf16(alo, bhi, acc[t]);
    }
  }
  // D layout: VGPR r -> (M = r + 8*half, N = l15 [+16 per tile])
  float* drow = D + (size_t)row0 * NCOLS;
  #pragma unroll
  for (int t = 0; t < TILES; ++t) {
    #pragma unroll
    for (int r = 0; r < 8; ++r) {
      int m = r + 8 * half;
      drow[(size_t)m * NCOLS + t * 16 + l15] = acc[t][r];
    }
  }
}

// ---------------- GAT edge pipeline ----------------

template <int C>
__global__ void node_logits(const float* __restrict__ h, const float* __restrict__ asrc,
                            const float* __restrict__ adst, float* __restrict__ als,
                            float* __restrict__ ald) {
  int n = blockIdx.x * blockDim.x + threadIdx.x;
  if (n >= N_NODES) return;
  const float* hr = h + (size_t)n * C;
  float s = 0.f, d = 0.f;
  #pragma unroll
  for (int c = 0; c < C; ++c) { float v = hr[c]; s += v * asrc[c]; d += v * adst[c]; }
  als[n] = s;
  ald[n] = d;
}

// pass 1: alpha = leaky_relu(al_src[s] + al_dst[d]) + log2(ew); segment max over dst
__global__ void edge_logits(const int* __restrict__ ei, const float* __restrict__ ew,
                            const float* __restrict__ als, const float* __restrict__ ald,
                            float* __restrict__ alpha, unsigned* __restrict__ amax) {
  long long e = blockIdx.x * (long long)blockDim.x + threadIdx.x;
  const long long total = (long long)E_EDGES + N_NODES;
  if (e >= total) return;
  int s, d; float lg;
  if (e < E_EDGES) { s = ei[e]; d = ei[E_EDGES + e]; lg = __log2f(ew[e]); }
  else             { s = d = (int)(e - E_EDGES);      lg = 0.f; }           // self-loop, w=1
  float z = als[s] + ald[d];
  z = (z > 0.f) ? z : NEG_SLOPE * z;
  float a = z + lg;
  alpha[e] = a;
  atomicMax(&amax[d], f2ord(a));
}

// pass 2: ex = exp(alpha - amax[d]); segment sum over dst; ex overwrites alpha
__global__ void edge_exp(const int* __restrict__ ei, float* __restrict__ alpha,
                         const unsigned* __restrict__ amax, float* __restrict__ denom) {
  long long e = blockIdx.x * (long long)blockDim.x + threadIdx.x;
  const long long total = (long long)E_EDGES + N_NODES;
  if (e >= total) return;
  int d = (e < E_EDGES) ? ei[E_EDGES + e] : (int)(e - E_EDGES);
  float ex = __expf(alpha[e] - ord2f(amax[d]));
  alpha[e] = ex;
  atomic_add_f32(&denom[d], ex);
}

// pass 3: agg[d] += h[s] * ex/denom[d]; lane = channel -> consecutive atomic addresses
template <int CLOG2>
__global__ void edge_aggregate(const int* __restrict__ ei, const float* __restrict__ h,
                               const float* __restrict__ ex, const float* __restrict__ denom,
                               float* __restrict__ agg) {
  constexpr int C = 1 << CLOG2;
  const long long total = (long long)E_EDGES + N_NODES;
  long long g = (blockIdx.x * (long long)blockDim.x + threadIdx.x) >> CLOG2;
  long long gs = ((long long)blockDim.x * gridDim.x) >> CLOG2;
  int ch = threadIdx.x & (C - 1);
  for (; g < total; g += gs) {
    int s, d;
    if (g < E_EDGES) { s = ei[g]; d = ei[E_EDGES + g]; }
    else             { s = d = (int)(g - E_EDGES); }
    float attn = ex[g] / denom[d];
    atomic_add_f32(&agg[(size_t)d * C + ch], h[(size_t)s * C + ch] * attn);
  }
}

__global__ void bias_relu(float* __restrict__ a, const float* __restrict__ b) {
  long long i = blockIdx.x * (long long)blockDim.x + threadIdx.x;
  if (i >= (long long)N_NODES * H1C) return;
  float v = a[i] + b[(int)(i & (H1C - 1))];
  a[i] = v > 0.f ? v : 0.f;
}

__global__ void bias_add(float* __restrict__ a, const float* __restrict__ b) {
  long long i = blockIdx.x * (long long)blockDim.x + threadIdx.x;
  if (i >= (long long)N_NODES * C_OUT) return;
  a[i] += b[(int)(i & (C_OUT - 1))];
}

// ---------------- host orchestration ----------------

extern "C" void kernel_launch(void* const* d_in, const int* in_sizes, int n_in,
                              void* d_out, int out_size, void* d_ws, size_t ws_size,
                              hipStream_t stream) {
  (void)in_sizes; (void)n_in; (void)out_size; (void)ws_size;
  const float* x   = (const float*)d_in[0];
  const int*   ei  = (const int*)d_in[1];
  const float* ew  = (const float*)d_in[2];
  const float* W1  = (const float*)d_in[3];
  const float* as1 = (const float*)d_in[4];
  const float* ad1 = (const float*)d_in[5];
  const float* b1  = (const float*)d_in[6];
  const float* W2  = (const float*)d_in[7];
  const float* as2 = (const float*)d_in[8];
  const float* ad2 = (const float*)d_in[9];
  const float* b2  = (const float*)d_in[10];
  float* out = (float*)d_out;

  // workspace layout (floats); all sub-offsets are 32B-aligned for v16bf loads
  float* ws    = (float*)d_ws;
  float* h1    = ws;                                     // N*32
  float* agg1  = h1   + (size_t)N_NODES * H1C;           // N*32 (becomes h1r)
  float* h2    = agg1 + (size_t)N_NODES * H1C;           // N*16
  float* alpha = h2   + (size_t)N_NODES * C_OUT;         // E+N
  float* als   = alpha + (size_t)(E_EDGES + N_NODES);    // N
  float* ald   = als + N_NODES;                          // N
  unsigned* amax = (unsigned*)(ald + N_NODES);           // N
  float* denom = (float*)(amax + N_NODES);               // N
  bf16_t* w1hi = (bf16_t*)(denom + N_NODES);             // 16*2*32*16 bf16
  bf16_t* w1lo = w1hi + 16 * 2 * 32 * 16;
  bf16_t* w2hi = w1lo + 16 * 2 * 32 * 16;                // 1*1*32*16 bf16
  bf16_t* w2lo = w2hi + 32 * 16;

  const long long total = (long long)E_EDGES + N_NODES;
  const int ET = 256;
  const int EB = (int)((total + ET - 1) / ET);

  pack_weights<<<1, 1024, 0, stream>>>(W1, W2, w1hi, w1lo, w2hi, w2lo);

  // deterministic re-init every call (graph replay safe)
  fill_u32<<<2048, 256, 0, stream>>>((unsigned*)agg1, 0u, (long long)N_NODES * H1C);
  fill_u32<<<2048, 256, 0, stream>>>((unsigned*)out, 0u, (long long)N_NODES * C_OUT);
  fill_u32<<<512, 256, 0, stream>>>(amax, ORD_NEG_INF, N_NODES);
  fill_u32<<<512, 256, 0, stream>>>((unsigned*)denom, 0u, N_NODES);

  // ---- layer 1 ----
  gemm_wmma<F_IN, H1C><<<N_NODES / 32, 64, 0, stream>>>(x, w1hi, w1lo, h1);
  node_logits<H1C><<<(N_NODES + 255) / 256, 256, 0, stream>>>(h1, as1, ad1, als, ald);
  edge_logits<<<EB, ET, 0, stream>>>(ei, ew, als, ald, alpha, amax);
  edge_exp<<<EB, ET, 0, stream>>>(ei, alpha, amax, denom);
  edge_aggregate<5><<<4096, 256, 0, stream>>>(ei, h1, alpha, denom, agg1);
  bias_relu<<<((long long)N_NODES * H1C + 255) / 256, 256, 0, stream>>>(agg1, b1);

  fill_u32<<<512, 256, 0, stream>>>(amax, ORD_NEG_INF, N_NODES);
  fill_u32<<<512, 256, 0, stream>>>((unsigned*)denom, 0u, N_NODES);

  // ---- layer 2 ----
  gemm_wmma<H1C, C_OUT><<<N_NODES / 32, 64, 0, stream>>>(agg1, w2hi, w2lo, h2);
  node_logits<C_OUT><<<(N_NODES + 255) / 256, 256, 0, stream>>>(h2, as2, ad2, als, ald);
  edge_logits<<<EB, ET, 0, stream>>>(ei, ew, als, ald, alpha, amax);
  edge_exp<<<EB, ET, 0, stream>>>(ei, alpha, amax, denom);
  edge_aggregate<4><<<4096, 256, 0, stream>>>(ei, h2, alpha, denom, out);
  bias_add<<<((long long)N_NODES * C_OUT + 255) / 256, 256, 0, stream>>>(out, b2);
}
